// M2M_4604204941663
// MI455X (gfx1250) — matrix operand order repaired and checked by reading, not compiled, not run on hardware
//
#include <hip/hip_runtime.h>

// ---------------------------------------------------------------------------
// Problem constants (match reference)
// ---------------------------------------------------------------------------
#define NN 100000   // nodes
#define CC 128      // channels
#define SS 6        // scales
#define EE 200000   // edges per pre/suc scale
#define ELR 50000   // left/right edges
#define LL 4        // layers

typedef __bf16 bf16_t;
typedef __attribute__((ext_vector_type(16))) __bf16 v16bf;
typedef __attribute__((ext_vector_type(8)))  __bf16 v8bf;
typedef __attribute__((ext_vector_type(8)))  float  v8f;

union BF16x16 { v16bf v; v8bf h[2]; };

__device__ __forceinline__ v8bf load8bf(const bf16_t* p) {
    return *(const v8bf*)p;   // 16-byte aligned by construction
}

// Stage one 128x128 bf16 weight matrix (32 KB) into LDS with the CDNA5
// async global->LDS DMA path. 256 threads x 128 B each. INST_OFFSET applies
// to both the LDS and global side, so one address pair covers 8 x b128.
__device__ __forceinline__ void stage_W_async(const bf16_t* __restrict__ Wg,
                                              bf16_t* Wl) {
    const bf16_t* src = Wg + (size_t)threadIdx.x * 64;          // 128 B / thread
    unsigned dst = (unsigned)(size_t)(Wl + (size_t)threadIdx.x * 64);
    asm volatile(
        "global_load_async_to_lds_b128 %0, %1, off\n\t"
        "global_load_async_to_lds_b128 %0, %1, off offset:16\n\t"
        "global_load_async_to_lds_b128 %0, %1, off offset:32\n\t"
        "global_load_async_to_lds_b128 %0, %1, off offset:48\n\t"
        "global_load_async_to_lds_b128 %0, %1, off offset:64\n\t"
        "global_load_async_to_lds_b128 %0, %1, off offset:80\n\t"
        "global_load_async_to_lds_b128 %0, %1, off offset:96\n\t"
        "global_load_async_to_lds_b128 %0, %1, off offset:112"
        :: "v"(dst), "v"(src) : "memory");
    asm volatile("s_wait_asynccnt 0x0" ::: "memory");
    __syncthreads();
}

// ---------------------------------------------------------------------------
// f32 -> bf16 conversion (feat + weights)
// ---------------------------------------------------------------------------
__global__ __launch_bounds__(256) void cvt_f32_bf16(const float* __restrict__ in,
                                                    bf16_t* __restrict__ out, int n) {
    int i = blockIdx.x * blockDim.x + threadIdx.x;
    if (i < n) out[i] = (bf16_t)in[i];
}

// ---------------------------------------------------------------------------
// Dense GEMM: out[m,n] = sum_k X[m,k] * W[n,k]   (X: rows x 128 bf16,
// W: 128x128 bf16 row-major staged in LDS, out: rows x 128 f32, plain stores)
// One wave per 16-row M-tile, grid-stride over tiles.
// ---------------------------------------------------------------------------
__global__ __launch_bounds__(256) void gemm_xwt(const bf16_t* __restrict__ X,
                                                const bf16_t* __restrict__ W,
                                                float* __restrict__ out, int nrows) {
    __shared__ bf16_t Wl[CC * CC];
    stage_W_async(W, Wl);

    const int lane = threadIdx.x & 31;
    const int m  = lane & 15;    // row within tile / output column within n-tile
    const int kh = lane >> 4;    // K half select
    const int mtiles = nrows >> 4;
    const int wavei  = blockIdx.x * 8 + (threadIdx.x >> 5);
    const int nw     = gridDim.x * 8;

    for (int mt = wavei; mt < mtiles; mt += nw) {
        const bf16_t* xr = X + (size_t)(mt * 16 + m) * CC + kh * 8;
        BF16x16 A[4];
#pragma unroll
        for (int kc = 0; kc < 4; ++kc) {
            A[kc].h[0] = load8bf(xr + kc * 32);        // K = kc*32 + kh*8 .. +7
            A[kc].h[1] = load8bf(xr + kc * 32 + 16);   // K = kc*32 + 16 + kh*8 .. +7
        }
#pragma unroll
        for (int nt = 0; nt < 8; ++nt) {
            const bf16_t* wr = Wl + (size_t)(nt * 16 + m) * CC + kh * 8;
            v8f acc = {};
#pragma unroll
            for (int kc = 0; kc < 4; ++kc) {
                BF16x16 B;
                B.h[0] = *(const v8bf*)(wr + kc * 32);      // ds_load_b128
                B.h[1] = *(const v8bf*)(wr + kc * 32 + 16);
                acc = __builtin_amdgcn_wmma_f32_16x16x32_bf16(
                    false, A[kc].v, false, B.v, (short)0, acc, false, false);
            }
#pragma unroll
            for (int j = 0; j < 8; ++j)
                out[(size_t)(mt * 16 + j + kh * 8) * CC + nt * 16 + m] = acc[j];
        }
    }
}

// ---------------------------------------------------------------------------
// Edge message kernel: for each 16-edge block of scale blockIdx.y,
//   temp[u] += feat_bf16[v] @ W_s^T   (f32 atomics into L2-resident temp)
// W_s staged in LDS once per workgroup; waves grid-stride over edge blocks.
// ---------------------------------------------------------------------------
__global__ __launch_bounds__(256) void edge_msg(const bf16_t* __restrict__ featb,
                                                const int* __restrict__ vidx,
                                                const int* __restrict__ uidx,
                                                const bf16_t* __restrict__ Wbase,
                                                float* __restrict__ temp,
                                                int edges_per_scale) {
    __shared__ bf16_t Wl[CC * CC];
    const int s = blockIdx.y;
    stage_W_async(Wbase + (size_t)s * CC * CC, Wl);

    const int lane = threadIdx.x & 31;
    const int m  = lane & 15;
    const int kh = lane >> 4;
    const int bps   = edges_per_scale >> 4;
    const int wavei = blockIdx.x * 8 + (threadIdx.x >> 5);
    const int nw    = gridDim.x * 8;
    const int* vbase = vidx + (size_t)s * edges_per_scale;
    const int* ubase = uidx + (size_t)s * edges_per_scale;

    for (int blk = wavei; blk < bps; blk += nw) {
        const int* v = vbase + blk * 16;
        const int* u = ubase + blk * 16;

        const bf16_t* ar = featb + (size_t)v[m] * CC + kh * 8;   // gathered row
        BF16x16 A[4];
#pragma unroll
        for (int kc = 0; kc < 4; ++kc) {
            A[kc].h[0] = load8bf(ar + kc * 32);
            A[kc].h[1] = load8bf(ar + kc * 32 + 16);
        }
        int urow[8];
#pragma unroll
        for (int j = 0; j < 8; ++j) urow[j] = u[j + kh * 8];     // scatter rows

#pragma unroll
        for (int nt = 0; nt < 8; ++nt) {
            const bf16_t* wr = Wl + (size_t)(nt * 16 + m) * CC + kh * 8;
            v8f acc = {};
#pragma unroll
            for (int kc = 0; kc < 4; ++kc) {
                BF16x16 B;
                B.h[0] = *(const v8bf*)(wr + kc * 32);          // ds_load_b128
                B.h[1] = *(const v8bf*)(wr + kc * 32 + 16);
                acc = __builtin_amdgcn_wmma_f32_16x16x32_bf16(
                    false, A[kc].v, false, B.v, (short)0, acc, false, false);
            }
#pragma unroll
            for (int j = 0; j < 8; ++j)
                atomicAdd(temp + (size_t)urow[j] * CC + nt * 16 + m, acc[j]);
        }
    }
}

// ---------------------------------------------------------------------------
// GroupNorm(1 group) -> *w + b -> relu -> bf16      (one wave per row)
// ---------------------------------------------------------------------------
__global__ __launch_bounds__(256) void gn_relu_bf16(const float* __restrict__ x,
                                                    const float* __restrict__ w,
                                                    const float* __restrict__ b,
                                                    bf16_t* __restrict__ out, int nrows) {
    int row  = (blockIdx.x * blockDim.x + threadIdx.x) >> 5;
    int lane = threadIdx.x & 31;
    if (row >= nrows) return;
    const float4 xv = *(const float4*)(x + (size_t)row * CC + lane * 4);
    float s  = xv.x + xv.y + xv.z + xv.w;
    float ss = xv.x * xv.x + xv.y * xv.y + xv.z * xv.z + xv.w * xv.w;
#pragma unroll
    for (int o = 16; o > 0; o >>= 1) {
        s  += __shfl_xor(s, o, 32);
        ss += __shfl_xor(ss, o, 32);
    }
    float mean = s * (1.0f / CC);
    float rs   = rsqrtf(ss * (1.0f / CC) - mean * mean + 1e-5f);
    float xe[4] = {xv.x, xv.y, xv.z, xv.w};
    bf16_t o4[4];
#pragma unroll
    for (int c = 0; c < 4; ++c) {
        int ch = lane * 4 + c;
        float y = (xe[c] - mean) * rs * w[ch] + b[ch];
        o4[c] = (bf16_t)(y > 0.0f ? y : 0.0f);
    }
    *(uint2*)(out + (size_t)row * CC + lane * 4) = *(const uint2*)o4;
}

// ---------------------------------------------------------------------------
// GroupNorm(y) -> *w + b -> + feat(residual) -> relu; write feat f32 + bf16.
// In-place on feat is safe (pure per-row read-before-write).
// ---------------------------------------------------------------------------
__global__ __launch_bounds__(256) void gn_res_relu(const float* __restrict__ y,
                                                   const float* __restrict__ w,
                                                   const float* __restrict__ b,
                                                   float* __restrict__ feat,
                                                   bf16_t* __restrict__ featb, int nrows) {
    int row  = (blockIdx.x * blockDim.x + threadIdx.x) >> 5;
    int lane = threadIdx.x & 31;
    if (row >= nrows) return;
    const float4 yv = *(const float4*)(y + (size_t)row * CC + lane * 4);
    float s  = yv.x + yv.y + yv.z + yv.w;
    float ss = yv.x * yv.x + yv.y * yv.y + yv.z * yv.z + yv.w * yv.w;
#pragma unroll
    for (int o = 16; o > 0; o >>= 1) {
        s  += __shfl_xor(s, o, 32);
        ss += __shfl_xor(ss, o, 32);
    }
    float mean = s * (1.0f / CC);
    float rs   = rsqrtf(ss * (1.0f / CC) - mean * mean + 1e-5f);
    float ye[4] = {yv.x, yv.y, yv.z, yv.w};
    float4 res = *(const float4*)(feat + (size_t)row * CC + lane * 4);
    float re[4] = {res.x, res.y, res.z, res.w};
    float  f4[4];
    bf16_t o4[4];
#pragma unroll
    for (int c = 0; c < 4; ++c) {
        int ch = lane * 4 + c;
        float v = (ye[c] - mean) * rs * w[ch] + b[ch] + re[c];
        v = v > 0.0f ? v : 0.0f;
        f4[c] = v;
        o4[c] = (bf16_t)v;
    }
    *(float4*)(feat + (size_t)row * CC + lane * 4) = *(const float4*)f4;
    *(uint2*)(featb + (size_t)row * CC + lane * 4) = *(const uint2*)o4;
}

// ---------------------------------------------------------------------------
// Host launcher
// ---------------------------------------------------------------------------
extern "C" void kernel_launch(void* const* d_in, const int* in_sizes, int n_in,
                              void* d_out, int out_size, void* d_ws, size_t ws_size,
                              hipStream_t stream) {
    (void)in_sizes; (void)n_in; (void)out_size; (void)ws_size;

    const float* feat_in = (const float*)d_in[0];
    const int* pre_u   = (const int*)d_in[1];
    const int* pre_v   = (const int*)d_in[2];
    const int* suc_u   = (const int*)d_in[3];
    const int* suc_v   = (const int*)d_in[4];
    const int* left_u  = (const int*)d_in[5];
    const int* left_v  = (const int*)d_in[6];
    const int* right_u = (const int*)d_in[7];
    const int* right_v = (const int*)d_in[8];
    const float* W_ctr   = (const float*)d_in[9];
    const float* norm_w  = (const float*)d_in[10];
    const float* norm_b  = (const float*)d_in[11];
    const float* W_ctr2  = (const float*)d_in[12];
    const float* ctr2_w  = (const float*)d_in[13];
    const float* ctr2_b  = (const float*)d_in[14];
    const float* W_left  = (const float*)d_in[15];
    const float* W_right = (const float*)d_in[16];
    const float* W_pre   = (const float*)d_in[17];
    const float* W_suc   = (const float*)d_in[18];

    // Workspace carve-up (all 256B-aligned sizes)
    char* ws = (char*)d_ws;
    float*  temp   = (float*)ws;                 ws += (size_t)NN * CC * 4;  // 51.2 MB
    bf16_t* featb  = (bf16_t*)ws;                ws += (size_t)NN * CC * 2;  // 25.6 MB
    bf16_t* xnorm  = (bf16_t*)ws;                ws += (size_t)NN * CC * 2;  // 25.6 MB
    bf16_t* Wb_ctr   = (bf16_t*)ws;              ws += (size_t)LL * CC * CC * 2;
    bf16_t* Wb_ctr2  = (bf16_t*)ws;              ws += (size_t)LL * CC * CC * 2;
    bf16_t* Wb_left  = (bf16_t*)ws;              ws += (size_t)LL * CC * CC * 2;
    bf16_t* Wb_right = (bf16_t*)ws;              ws += (size_t)LL * CC * CC * 2;
    bf16_t* Wb_pre   = (bf16_t*)ws;              ws += (size_t)LL * SS * CC * CC * 2;
    bf16_t* Wb_suc   = (bf16_t*)ws;              ws += (size_t)LL * SS * CC * CC * 2;

    float* feat = (float*)d_out;   // f32 feat lives in d_out (final layer leaves answer)

    // feat (f32) into d_out; bf16 shadow of feat
    hipMemcpyAsync(feat, feat_in, (size_t)NN * CC * sizeof(float),
                   hipMemcpyDeviceToDevice, stream);
    {
        int n = NN * CC;
        cvt_f32_bf16<<<(n + 255) / 256, 256, 0, stream>>>(feat_in, featb, n);
    }
    // Weights -> bf16 (cheap; redone every call for determinism)
    {
        int nw = LL * CC * CC;
        cvt_f32_bf16<<<(nw + 255) / 256, 256, 0, stream>>>(W_ctr,   Wb_ctr,   nw);
        cvt_f32_bf16<<<(nw + 255) / 256, 256, 0, stream>>>(W_ctr2,  Wb_ctr2,  nw);
        cvt_f32_bf16<<<(nw + 255) / 256, 256, 0, stream>>>(W_left,  Wb_left,  nw);
        cvt_f32_bf16<<<(nw + 255) / 256, 256, 0, stream>>>(W_right, Wb_right, nw);
        int ns = LL * SS * CC * CC;
        cvt_f32_bf16<<<(ns + 255) / 256, 256, 0, stream>>>(W_pre, Wb_pre, ns);
        cvt_f32_bf16<<<(ns + 255) / 256, 256, 0, stream>>>(W_suc, Wb_suc, ns);
    }

    // Grid sizing: ~2 M-tiles per wave (GEMM), ~4 edge blocks per wave (edges)
    const dim3 gemm_grid((NN / 16 / 2 + 7) / 8);          // 391 WGs
    const dim3 ps_grid((EE / 16 / 4 + 7) / 8, SS);        // 391 x 6 WGs
    const dim3 lr_grid((ELR / 16 / 4 + 7) / 8, 1);        // 98 WGs
    const int  gn_blks = (NN * 32 + 255) / 256;

    for (int l = 0; l < LL; ++l) {
        const size_t wcc = (size_t)l * CC * CC;
        const size_t wsc = (size_t)l * SS * CC * CC;

        // temp = feat @ W_ctr^T  (plain stores initialize temp)
        gemm_xwt<<<gemm_grid, 256, 0, stream>>>(featb, Wb_ctr + wcc, temp, NN);

        // scatter-add edge messages into temp (f32 atomics, L2-resident)
        edge_msg<<<ps_grid, 256, 0, stream>>>(featb, pre_v, pre_u,
                                              Wb_pre + wsc, temp, EE);
        edge_msg<<<ps_grid, 256, 0, stream>>>(featb, suc_v, suc_u,
                                              Wb_suc + wsc, temp, EE);
        edge_msg<<<lr_grid, 256, 0, stream>>>(featb, left_v, left_u,
                                              Wb_left + wcc, temp, ELR);
        edge_msg<<<lr_grid, 256, 0, stream>>>(featb, right_v, right_u,
                                              Wb_right + wcc, temp, ELR);

        // xnorm = bf16(relu(GN(temp)*w+b))
        gn_relu_bf16<<<gn_blks, 256, 0, stream>>>(temp, norm_w + (size_t)l * CC,
                                                  norm_b + (size_t)l * CC, xnorm, NN);
        // temp = xnorm @ W_ctr2^T   (temp reused as y)
        gemm_xwt<<<gemm_grid, 256, 0, stream>>>(xnorm, Wb_ctr2 + wcc, temp, NN);
        // feat = relu(GN(temp)*w+b + feat); refresh bf16 shadow
        gn_res_relu<<<gn_blks, 256, 0, stream>>>(temp, ctr2_w + (size_t)l * CC,
                                                 ctr2_b + (size_t)l * CC,
                                                 feat, featb, NN);
    }
}
